// AsymmetricLossCustomPriorityRankNewNeg_18064632447150
// MI455X (gfx1250) — matrix operand order repaired and checked by reading, not compiled, not run on hardware
//
#include <hip/hip_runtime.h>
#include <hip/hip_bf16.h>
#include <math.h>

typedef __attribute__((ext_vector_type(16))) _Float16 v16h;
typedef __attribute__((ext_vector_type(8)))  float    v8f;

#define B_ROWS   2048
#define C_COLS   9605
#define KFULL    9600            // (C_COLS / 32) * 32 : last full 32-wide K block
#define NCH32    301             // ceil(C_COLS / 32)
#define NLAB     8
#define NTHREADS 256
#define CSPLIT   8
#define CHUNK    1216            // ceil(C/CSPLIT) rounded to multiple of 32

// ---------------------------------------------------------------- helpers

__device__ __forceinline__ float sigmoid_f(float v) {
    return 1.0f / (1.0f + expf(-v));
}

__device__ __forceinline__ float block_reduce_max(float v, float* red, int tid) {
    red[tid] = v; __syncthreads();
    #pragma unroll
    for (int s = NTHREADS / 2; s > 0; s >>= 1) {
        if (tid < s) red[tid] = fmaxf(red[tid], red[tid + s]);
        __syncthreads();
    }
    float r = red[0]; __syncthreads();
    return r;
}

__device__ __forceinline__ float block_reduce_sum(float v, float* red, int tid) {
    red[tid] = v; __syncthreads();
    #pragma unroll
    for (int s = NTHREADS / 2; s > 0; s >>= 1) {
        if (tid < s) red[tid] = red[tid] + red[tid + s];
        __syncthreads();
    }
    float r = red[0]; __syncthreads();
    return r;
}

// ------------------------------------------------ 1) pack wl bools -> bitmask word per class
__global__ void build_mask_kernel(const unsigned char* __restrict__ wl,
                                  unsigned int* __restrict__ maskWord) {
    int c = blockIdx.x * blockDim.x + threadIdx.x;
    if (c < C_COLS) {
        unsigned int mw = 0u;
        #pragma unroll
        for (int l = 0; l < NLAB; ++l)
            if (wl[l * C_COLS + c]) mw |= (1u << l);
        maskWord[c] = mw;
    }
}

// ------------------------------------------------ 1b) prepack B tiles (wl^T) in WMMA lane layout
// bpack[c32 * 32 + lane] = the 16 f16 values lane `lane` feeds as SRC1 for K-block c32.
// B-matrix 32x16 f16 layout: elements 0..7 hold K=8*half+e, elements 8..15 hold K=16+8*half+(e-8) (matches A);
// column n = lane & 15 (labels 8..15 are zero padding). Zero beyond C.
__global__ void build_bpack_kernel(const unsigned int* __restrict__ maskWord,
                                   v16h* __restrict__ bpack) {
    int idx = blockIdx.x * blockDim.x + threadIdx.x;   // [0, NCH32*32)
    if (idx < NCH32 * 32) {
        int c32  = idx >> 5;
        int lane = idx & 31;
        int half = lane >> 4;
        int n    = lane & 15;
        v16h bv;
        #pragma unroll
        for (int e = 0; e < 16; ++e) {
            int k = c32 * 32 + ((e < 8) ? (half * 8 + e) : (16 + half * 8 + (e - 8)));
            unsigned int mw = (k < C_COLS) ? maskWord[k] : 0u;
            bv[e] = ((n < NLAB) && ((mw >> n) & 1u)) ? (_Float16)1.0f : (_Float16)0.0f;
        }
        bpack[idx] = bv;
    }
}

// ------------------------------------------------ 2) counts[chunk][b][l] = sum_c y[b,c]*wl[l,c]
// One wave32 per (16-row group, C-chunk). V_WMMA_F32_16X16X32_F16, K=32 per step.
// Main loop is bounds-check free (full 32-wide K blocks only); tail peeled.
__global__ __launch_bounds__(32)
void haspos_wmma_kernel(const float* __restrict__ y,
                        const v16h* __restrict__ bpack,
                        float* __restrict__ counts) {
    const int t       = threadIdx.x;   // 0..31
    const int half    = t >> 4;        // 0: lanes 0-15, 1: lanes 16-31
    const int m       = t & 15;        // row (A) / column-label (B) / N (C)
    const int rowBase = blockIdx.x * 16;
    const int kStart  = blockIdx.y * CHUNK;
    int kEndMain = kStart + CHUNK;
    if (kEndMain > KFULL) kEndMain = KFULL;

    const float* yrow = y + (size_t)(rowBase + m) * C_COLS;

    v8f acc = {};
    for (int k0 = kStart; k0 < kEndMain; k0 += 32) {
        // A-matrix 16x32 f16: lane half -> K (0..7 | 8..15) in V0-3 and (16..23 | 24..31) in V4-7
        const float* pA = yrow + k0 + half * 8;
        v16h a;
        #pragma unroll
        for (int e = 0; e < 8; ++e) {
            a[e]     = (_Float16)pA[e];        // elements 0..7  (VGPR0-3)
            a[8 + e] = (_Float16)pA[16 + e];   // elements 8..15 (VGPR4-7)
        }
        v16h b = bpack[(k0 >> 5) * 32 + t];    // 32B aligned, coalesced
        acc = __builtin_amdgcn_wmma_f32_16x16x32_f16(
            /*neg_a=*/false, a, /*neg_b=*/false, b,
            /*c_mod=*/(short)0, acc, /*reuse_a=*/false, /*reuse_b=*/false);
    }

    // peeled tail (K block [9600, 9632); valid K < 9605) — only last chunk owns it
    if (blockIdx.y == CSPLIT - 1) {
        const int k0 = KFULL;
        v16h a;
        #pragma unroll
        for (int e = 0; e < 8; ++e) {
            int kA0 = k0 + half * 8 + e;
            int kA1 = k0 + 16 + half * 8 + e;
            a[e]     = (kA0 < C_COLS) ? (_Float16)yrow[kA0] : (_Float16)0.0f;
            a[8 + e] = (kA1 < C_COLS) ? (_Float16)yrow[kA1] : (_Float16)0.0f;
        }
        v16h b = bpack[(k0 >> 5) * 32 + t];    // zero-padded beyond C
        acc = __builtin_amdgcn_wmma_f32_16x16x32_f16(
            false, a, false, b, (short)0, acc, false, false);
    }

    // C/D f32 16x16 layout: VGPR v -> M = v + 8*half, N = m
    float* cbase = counts + ((size_t)blockIdx.y * B_ROWS) * 16;
    #pragma unroll
    for (int v = 0; v < 8; ++v) {
        int row = rowBase + v + half * 8;
        cbase[row * 16 + m] = acc[v];
    }
}

// ------------------------------------------------ 3) fused per-row pass over x
__global__ __launch_bounds__(NTHREADS)
void row_loss_kernel(const float* __restrict__ x,
                     const unsigned int* __restrict__ maskWord,
                     const float* __restrict__ counts,
                     float* __restrict__ row_out) {
    __shared__ float red[NTHREADS];

    const int b   = blockIdx.x;
    const int tid = threadIdx.x;
    const float* xrow = x + (size_t)b * C_COLS;

    float t11[11];
    #pragma unroll
    for (int i = 0; i < 11; ++i) t11[i] = -INFINITY;
    float lmax[NLAB];
    #pragma unroll
    for (int l = 0; l < NLAB; ++l) lmax[l] = -INFINITY;
    float umax = -INFINITY;

    for (int c = tid; c < C_COLS; c += NTHREADS) {
        __builtin_prefetch(xrow + c + 4 * NTHREADS, 0, 0);
        float s = sigmoid_f(xrow[c]);
        unsigned int mw = maskWord[c];
        if (mw) {
            umax = fmaxf(umax, s);
            #pragma unroll
            for (int l = 0; l < NLAB; ++l)
                if ((mw >> l) & 1u) lmax[l] = fmaxf(lmax[l], s);
        }
        // sorted-descending insertion into per-thread top-11 (registers only)
        if (s > t11[10]) {
            t11[10] = s;
            #pragma unroll
            for (int i = 10; i > 0; --i) {
                float hi = fmaxf(t11[i - 1], t11[i]);
                float lo = fminf(t11[i - 1], t11[i]);
                t11[i - 1] = hi; t11[i] = lo;
            }
        }
    }

    // block-wide per-label maxes + union max
    float gl[NLAB];
    #pragma unroll
    for (int l = 0; l < NLAB; ++l) gl[l] = block_reduce_max(lmax[l], red, tid);
    float g_union = block_reduce_max(umax, red, tid);

    // 11th-largest across the block: count-based selection (tie-safe like top_k)
    float bound = INFINITY;
    float kleft = 11.0f;
    float v11   = -INFINITY;
    bool  done  = false;                 // uniform across block (derived from broadcasts)
    for (int round = 0; round < 11; ++round) {
        float lm = -INFINITY;
        #pragma unroll
        for (int i = 0; i < 11; ++i) {
            float v = t11[i];
            if (v < bound) lm = fmaxf(lm, v);
        }
        float gm = block_reduce_max(done ? -INFINITY : lm, red, tid);
        float lc = 0.0f;
        #pragma unroll
        for (int i = 0; i < 11; ++i) lc += (t11[i] == gm) ? 1.0f : 0.0f;
        float gc = block_reduce_sum(done ? 0.0f : lc, red, tid);
        if (!done) {
            if (gc >= kleft) { v11 = gm; done = true; }
            else             { kleft -= gc; bound = gm; }
        }
    }
    float thres = fmaxf(v11, 0.5f);   // ALPHA_OTHER

    if (tid == 0) {
        float cmax = -INFINITY, imax = -INFINITY;
        bool anyc = false, anyi = false;
        #pragma unroll
        for (int l = 0; l < NLAB; ++l) {
            bool hp = false;
            #pragma unroll
            for (int ch = 0; ch < CSPLIT; ++ch)
                hp = hp || (counts[((size_t)ch * B_ROWS + b) * 16 + l] > 0.0f);
            if (hp) { anyc = true; cmax = fmaxf(cmax, gl[l]); }
            else    { anyi = true; imax = fmaxf(imax, gl[l]); }
        }
        float x1   = anyc ? cmax : thres;
        float x2   = anyc ? (anyi ? fmaxf(imax, thres) : thres) : g_union;
        float coef = anyc ? 1.0f : 0.5f;                 // 1 - ALPHA
        float d    = x2 - x1 + 0.1f;                      // ALPHA1
        float rank = ((d > 0.0f) ? 2.0f : 1.0f)           // ALPHA2
                     * sigmoid_f(10.0f * d);              // ALPHA3
        row_out[b] = coef * rank;
    }
}

// ------------------------------------------------ 4) deterministic mean
__global__ __launch_bounds__(NTHREADS)
void mean_kernel(const float* __restrict__ row_out, float* __restrict__ out) {
    __shared__ float red[NTHREADS];
    const int tid = threadIdx.x;
    float s = 0.0f;
    for (int i = tid; i < B_ROWS; i += NTHREADS) s += row_out[i];
    red[tid] = s; __syncthreads();
    #pragma unroll
    for (int st = NTHREADS / 2; st > 0; st >>= 1) {
        if (tid < st) red[tid] += red[tid + st];
        __syncthreads();
    }
    if (tid == 0) out[0] = red[0] / (float)B_ROWS;
}

// ---------------------------------------------------------------- launch

extern "C" void kernel_launch(void* const* d_in, const int* in_sizes, int n_in,
                              void* d_out, int out_size, void* d_ws, size_t ws_size,
                              hipStream_t stream) {
    (void)in_sizes; (void)n_in; (void)out_size; (void)ws_size;

    const float*         x  = (const float*)d_in[0];
    const float*         y  = (const float*)d_in[1];
    /* d_in[2] = y_neg: consumed by the reference but never affects the loss */
    const unsigned char* wl = (const unsigned char*)d_in[3];   // bool -> 1 byte/elem
    float* out = (float*)d_out;

    // workspace layout (bytes):
    //   [0,        38464)   maskWord : uint32[C]                (38420, padded)
    //   [38464,   346688)   bpack    : v16h[NCH32*32]           (308224, 32B-aligned)
    //   [346688, 1395264)   counts   : f32[CSPLIT][B][16]       (1048576)
    //   [1395264,1403456)   row_out  : f32[B]                   (8192)
    char* ws = (char*)d_ws;
    unsigned int* maskWord = (unsigned int*)ws;
    v16h*         bpack    = (v16h*)(ws + 38464);
    float*        counts   = (float*)(ws + 346688);
    float*        row_out  = (float*)(ws + 346688 + (size_t)CSPLIT * B_ROWS * 16 * 4);

    build_mask_kernel<<<(C_COLS + 255) / 256, 256, 0, stream>>>(wl, maskWord);
    build_bpack_kernel<<<(NCH32 * 32 + 255) / 256, 256, 0, stream>>>(maskWord, bpack);

    dim3 g2(B_ROWS / 16, CSPLIT);
    haspos_wmma_kernel<<<g2, 32, 0, stream>>>(y, bpack, counts);

    row_loss_kernel<<<B_ROWS, NTHREADS, 0, stream>>>(x, maskWord, counts, row_out);

    mean_kernel<<<1, NTHREADS, 0, stream>>>(row_out, out);
}
